// LocalMultiHeadAttention_6236292514265
// MI455X (gfx1250) — compile-verified, matrix operand order checked by reading, not run on hardware
//
#include <hip/hip_runtime.h>

typedef _Float16 half_t;
typedef __attribute__((ext_vector_type(16))) _Float16 v16h;
typedef __attribute__((ext_vector_type(8)))  _Float16 v8h;
typedef __attribute__((ext_vector_type(8)))  float    v8f;
typedef __attribute__((ext_vector_type(4)))  float    v4f;
typedef __attribute__((ext_vector_type(4)))  unsigned int v4u;
typedef __attribute__((ext_vector_type(8)))  int      v8i;
typedef __attribute__((ext_vector_type(4)))  int      v4i;

#define B_  4
#define L_  4096
#define D_  1024
#define H_  16
#define E_  64
#define W_  128
#define NW_ 32
static constexpr size_t HSZ = (size_t)B_ * H_ * L_ * E_;  // halves per Q/K/V tensor

__device__ __forceinline__ v16h mk16(v8h lo, v8h hi) {
  v16h r;
#pragma unroll
  for (int i = 0; i < 8; ++i) { r[i] = lo[i]; r[i + 8] = hi[i]; }
  return r;
}

__device__ __forceinline__ v8f wmma16x16x32(v16h a, v16h b, v8f c) {
  // D = A(16x32 f16) x B(32x16 f16) + C(16x16 f32)
  return __builtin_amdgcn_wmma_f32_16x16x32_f16(false, a, false, b, (short)0, c,
                                                false, false);
}

__device__ __forceinline__ v8f zero8() {
  v8f z;
#pragma unroll
  for (int i = 0; i < 8; ++i) z[i] = 0.0f;
  return z;
}

// ---------------------------------------------------------------------------
// Tensor Data Mover: 2D tile load Global -> LDS with row padding.
// dims/tiles/stride in elements (data_size = 2B). padIv/padAmt per D# encoding:
// pad after (1<<padIv) 8-byte units, insert (padAmt+1) DWORDs.
// ---------------------------------------------------------------------------
__device__ __forceinline__ void tdm_load_2d_f16(unsigned int ldsAddr,
                                                const void* gptr,
                                                unsigned int dim0, unsigned int dim1,
                                                unsigned int tile0, unsigned int tile1,
                                                unsigned long long stride0,
                                                unsigned int padIv, unsigned int padAmt) {
  const unsigned long long ga = (unsigned long long)(uintptr_t)gptr;
  v4u g0;
  g0[0] = 1u;                                   // count=1 (valid user descriptor)
  g0[1] = ldsAddr;                              // LDS byte address
  g0[2] = (unsigned int)ga;                     // global_addr[31:0]
  g0[3] = (unsigned int)((ga >> 32) & 0x01FFFFFFu) | (2u << 30);  // [56:32], type=2
  v8i g1;
  g1[0] = (int)((1u << 16)                      // data_size = 2 bytes
                | (1u << 20)                    // pad_enable
                | (padIv << 22) | (padAmt << 25));
  g1[1] = (int)(dim0 << 16);                    // tensor_dim0 low16 @ bits 63:48
  g1[2] = (int)((dim0 >> 16) | (dim1 << 16));   // dim0 hi / dim1 lo
  g1[3] = (int)((dim1 >> 16) | (tile0 << 16));  // dim1 hi / tile_dim0
  g1[4] = (int)tile1;                           // tile_dim1 (tile_dim2 = 0)
  g1[5] = (int)(unsigned int)stride0;           // tensor_dim0_stride[31:0]
  g1[6] = (int)(unsigned int)(stride0 >> 32);   // stride0[47:32] (dim1_stride lo = 0)
  g1[7] = 0;
  const v4i z4 = {0, 0, 0, 0};                  // groups 2/3: dims 3/4 unused
  const v8i z8 = {0, 0, 0, 0, 0, 0, 0, 0};
  __builtin_amdgcn_tensor_load_to_lds(g0, g1, z4, z4, z8, 0);
}

// ---------------------------------------------------------------------------
// Kernel 1: qkv = x @ w_qkv  (f32 in, f16 out)
//   Q, K -> [b][h][l][e] f16 (Q pre-scaled by E^-0.5);  V -> [b][h][e][l] f16
//   All parts re-tiled through LDS so every global store is a 128B b128 run.
// ---------------------------------------------------------------------------
__global__ __launch_bounds__(256)
void qkv_gemm_f16(const float* __restrict__ x, const float* __restrict__ w,
                  half_t* __restrict__ qkv) {
  __shared__ __align__(16) half_t smem[2 * 128 * 72];
  half_t(*As)[72]  = (half_t(*)[72])smem;              // A tile 128x32 (padded)
  half_t(*BsT)[72] = (half_t(*)[72])(smem + 128 * 72); // B tile transposed [n][k]

  const int n0 = blockIdx.x * 128;
  const int m0 = blockIdx.y * 128;
  const int t = threadIdx.x;
  const int lane = t & 31;
  const int wave = t >> 5;
  const int wm = (wave & 3) * 32;   // wave's M offset in tile
  const int wn = (wave >> 2) * 64;  // wave's N offset in tile
  const int lr = lane & 15;
  const int kh = lane >> 4;

  v8f acc[2][4];
#pragma unroll
  for (int mt = 0; mt < 2; ++mt)
#pragma unroll
    for (int nt = 0; nt < 4; ++nt) acc[mt][nt] = zero8();

  const int ra = t >> 1, ha = t & 1;  // A staging: row, 16-col segment
  const int rb = t >> 3, sb = t & 7;  // B staging: k-row, 16-col segment

  for (int k0 = 0; k0 < D_; k0 += 32) {
    // stage A (f32 -> f16), 16B-aligned vector LDS stores
    {
      const float* src = x + (size_t)(m0 + ra) * D_ + k0 + ha * 16;
      __builtin_prefetch(src + 32, 0, 0);  // next k-tile -> global_prefetch_b8
#pragma unroll
      for (int q = 0; q < 2; ++q) {
        v4f f0 = *(const v4f*)(src + q * 8);
        v4f f1 = *(const v4f*)(src + q * 8 + 4);
        v8h hh;
#pragma unroll
        for (int j = 0; j < 4; ++j) { hh[j] = (half_t)f0[j]; hh[j + 4] = (half_t)f1[j]; }
        *(v8h*)&As[ra][ha * 16 + q * 8] = hh;
      }
    }
    // stage B transposed (coalesced global reads, u16 LDS scatter)
    {
      const float* src = w + (size_t)(k0 + rb) * (3 * D_) + n0 + sb * 16;
      __builtin_prefetch(src + (size_t)32 * (3 * D_), 0, 0);
#pragma unroll
      for (int q = 0; q < 4; ++q) {
        v4f f = *(const v4f*)(src + q * 4);
#pragma unroll
        for (int j = 0; j < 4; ++j) BsT[sb * 16 + q * 4 + j][rb] = (half_t)f[j];
      }
    }
    __syncthreads();

    // fragment loads per documented 16-bit A/B layouts (lane=row, kh=K-half)
    v16h aF[2];
#pragma unroll
    for (int mt = 0; mt < 2; ++mt) {
      const half_t* rp = &As[wm + 16 * mt + lr][0];
      aF[mt] = mk16(*(const v8h*)(rp + kh * 8), *(const v8h*)(rp + 16 + kh * 8));
    }
    v16h bF[4];
#pragma unroll
    for (int nt = 0; nt < 4; ++nt) {
      const half_t* rp = &BsT[wn + 16 * nt + lr][0];
      bF[nt] = mk16(*(const v8h*)(rp + kh * 8), *(const v8h*)(rp + 16 + kh * 8));
    }
#pragma unroll
    for (int mt = 0; mt < 2; ++mt)
#pragma unroll
      for (int nt = 0; nt < 4; ++nt)
        acc[mt][nt] = wmma16x16x32(aF[mt], bF[nt], acc[mt][nt]);

    __syncthreads();
  }

  // unified epilogue: dump accumulators into Cs[n_local][m_local] (m-contiguous
  // 16B stores), then part-specific fully-coalesced writeback.
  const int part = n0 >> 10;  // uniform per block (1024 % 128 == 0)
  const float scl = (part == 0) ? 0.125f : 1.0f;  // fold E^-0.5 into Q
  half_t(*Cs)[136] = (half_t(*)[136])smem;        // 272B rows (bank stride 4)
#pragma unroll
  for (int mt = 0; mt < 2; ++mt)
#pragma unroll
    for (int nt = 0; nt < 4; ++nt) {
      v8h hh;
#pragma unroll
      for (int i = 0; i < 8; ++i) hh[i] = (half_t)(acc[mt][nt][i] * scl);
      *(v8h*)&Cs[wn + 16 * nt + lr][wm + 16 * mt + 8 * kh] = hh;
    }
  __syncthreads();

  const int bb = m0 >> 12;
  const int sg = t & 1;
  if (part < 2) {
    // Q/K -> [b][h][l][e]: thread = (l-row, head-half); 128B contiguous stores
    const int ml = t >> 1;
    const int h0 = ((n0 & 1023) >> 6) + sg;
    const int l = (m0 & 4095) + ml;
    half_t* dst = qkv + (size_t)part * HSZ + (((size_t)bb * H_ + h0) * L_ + l) * E_;
#pragma unroll
    for (int u = 0; u < 8; ++u) {
      v8h hh;
#pragma unroll
      for (int j = 0; j < 8; ++j) hh[j] = Cs[sg * 64 + u * 8 + j][ml];
      *(v8h*)(dst + u * 8) = hh;
    }
  } else {
    // V -> [b][h][e][l]: thread = (e-row, l-half); 128B contiguous stores
    const int nl = t >> 1;
    const int gn = n0 + nl;
    const int cc = gn & 1023, hh2 = cc >> 6, e = cc & 63;
    const int lb = (m0 & 4095) + sg * 64;
    half_t* dst = qkv + 2 * HSZ + ((((size_t)bb * H_ + hh2) * E_ + e) * L_ + lb);
#pragma unroll
    for (int u = 0; u < 8; ++u)
      *(v8h*)(dst + u * 8) = *(const v8h*)&Cs[nl][sg * 64 + u * 8];
  }
}

// ---------------------------------------------------------------------------
// Kernel 2: local attention, one block per (win, h, b); 8 waves x 16 Q rows.
// Double-buffered TDM staging of K/V chunks overlaps DMA with WMMA + softmax.
// ---------------------------------------------------------------------------
__global__ __launch_bounds__(256)
void local_attn_f16(const half_t* __restrict__ qkv, float* __restrict__ out) {
  __shared__ __align__(16) half_t Ks[2][W_][72];       // K chunk row-major + pad
  __shared__ __align__(16) half_t VsT[2][E_][W_ + 8];  // V chunk [e][key] + pad
  __shared__ __align__(16) half_t Ps[8][16][32];       // per-wave P relay (D->A)

  const int win = blockIdx.x, h = blockIdx.y, b = blockIdx.z;
  const int t = threadIdx.x, lane = t & 31, wave = t >> 5;
  const int lr = lane & 15, kh = lane >> 4;

  const half_t* Qf = qkv;
  const half_t* Kf = qkv + HSZ;        // [b][h][l][e]
  const half_t* Vf = qkv + 2 * HSZ;    // [b][h][e][l]
  const size_t hb = (size_t)(b * H_ + h) * (size_t)(L_ * E_);
  const int l0 = win * W_;
  const int c0 = (win == 0) ? 1 : 0;        // skip fully-padded left window
  const int c1 = (win == NW_ - 1) ? 1 : 2;  // skip fully-padded right window

  const unsigned int ksLds[2] = {(unsigned int)(uintptr_t)&Ks[0][0][0],
                                 (unsigned int)(uintptr_t)&Ks[1][0][0]};
  const unsigned int vsLds[2] = {(unsigned int)(uintptr_t)&VsT[0][0][0],
                                 (unsigned int)(uintptr_t)&VsT[1][0][0]};

  auto issue_chunk = [&](int c) {
    const int lc = l0 + (c - 1) * W_;
    const int bi = c & 1;
    // K: 128 rows x 64 halves, row stride 64; LDS rows padded 64->72 halves
    tdm_load_2d_f16(ksLds[bi], Kf + hb + (size_t)lc * E_,
                    E_, W_, E_, W_, /*stride0=*/E_, /*padIv=*/4, /*padAmt=*/3);
    // V: 64 e-rows x 128 halves, row stride L; LDS rows padded 128->136 halves
    tdm_load_2d_f16(vsLds[bi], Vf + hb + (size_t)lc,
                    W_, E_, W_, E_, /*stride0=*/L_, /*padIv=*/5, /*padAmt=*/3);
  };

  if (wave == 0) issue_chunk(c0);  // prime the pipeline

  // per-wave Q strip (16 rows x 64), already scaled in the GEMM
  const half_t* qsrc = Qf + hb + (size_t)(l0 + wave * 16 + lr) * E_;
  v16h qF[2];
#pragma unroll
  for (int kt = 0; kt < 2; ++kt)
    qF[kt] = mk16(*(const v8h*)(qsrc + kt * 32 + kh * 8),
                  *(const v8h*)(qsrc + kt * 32 + 16 + kh * 8));

  float mrow[8], lrow[8];
  v8f o[4];
#pragma unroll
  for (int i = 0; i < 8; ++i) { mrow[i] = -3.0e38f; lrow[i] = 0.0f; }
#pragma unroll
  for (int e4 = 0; e4 < 4; ++e4) o[e4] = zero8();

  for (int c = c0; c <= c1; ++c) {
    if (wave == 0) {
      if (c < c1) {  // overlap next chunk's DMA with this chunk's compute
        issue_chunk(c + 1);
        __builtin_amdgcn_s_wait_tensorcnt(2);  // oldest pair (chunk c) retired
      } else {
        __builtin_amdgcn_s_wait_tensorcnt(0);
      }
    }
    __syncthreads();  // chunk c resident for all waves
    const half_t(*Kc)[72]     = Ks[c & 1];
    const half_t(*Vc)[W_ + 8] = VsT[c & 1];

    // S strip = Q (16x64) x Kc^T (64x128): 8 tiles x 2 k-steps
    v8f s[8];
#pragma unroll
    for (int nt = 0; nt < 8; ++nt) {
      s[nt] = zero8();
      const half_t* rp = &Kc[nt * 16 + lr][0];
#pragma unroll
      for (int kt = 0; kt < 2; ++kt)
        s[nt] = wmma16x16x32(qF[kt],
                             mk16(*(const v8h*)(rp + kt * 32 + kh * 8),
                                  *(const v8h*)(rp + kt * 32 + 16 + kh * 8)),
                             s[nt]);
    }

    // online softmax: row max via 16-lane butterfly (matches C/D row split)
    float mnew[8], corr[8], rsum[8];
#pragma unroll
    for (int i = 0; i < 8; ++i) {
      float mx = s[0][i];
#pragma unroll
      for (int nt = 1; nt < 8; ++nt) mx = fmaxf(mx, s[nt][i]);
#pragma unroll
      for (int d = 1; d < 16; d <<= 1) mx = fmaxf(mx, __shfl_xor(mx, d, 32));
      mnew[i] = fmaxf(mrow[i], mx);
      corr[i] = __expf(mrow[i] - mnew[i]);
      mrow[i] = mnew[i];
      rsum[i] = 0.0f;
    }
#pragma unroll
    for (int e4 = 0; e4 < 4; ++e4)
#pragma unroll
      for (int i = 0; i < 8; ++i) o[e4][i] *= corr[i];

    // O += P(16x128) x Vc(128x64), 32 keys per slab via per-wave LDS relay
#pragma unroll
    for (int kt = 0; kt < 4; ++kt) {
#pragma unroll
      for (int q = 0; q < 2; ++q) {
        const int nt = 2 * kt + q;
#pragma unroll
        for (int i = 0; i < 8; ++i) {
          const float p = __expf(s[nt][i] - mnew[i]);
          rsum[i] += p;
          Ps[wave][8 * kh + i][q * 16 + lr] = (half_t)p;  // D-layout dump
        }
      }
      const half_t* pp = &Ps[wave][lr][0];                 // A-layout pickup
      v16h pF = mk16(*(const v8h*)(pp + kh * 8), *(const v8h*)(pp + 16 + kh * 8));
#pragma unroll
      for (int e4 = 0; e4 < 4; ++e4) {
        const half_t* vp = &Vc[e4 * 16 + lr][0];
        o[e4] = wmma16x16x32(pF,
                             mk16(*(const v8h*)(vp + kt * 32 + kh * 8),
                                  *(const v8h*)(vp + kt * 32 + 16 + kh * 8)),
                             o[e4]);
      }
    }
#pragma unroll
    for (int i = 0; i < 8; ++i) {
      float rs = rsum[i];
#pragma unroll
      for (int d = 1; d < 16; d <<= 1) rs += __shfl_xor(rs, d, 32);
      lrow[i] = lrow[i] * corr[i] + rs;
    }
    __syncthreads();  // all waves done with buf[c&1] before it is re-filled
  }

  // normalize and store to (B, L, H*E) fp32
#pragma unroll
  for (int i = 0; i < 8; ++i) lrow[i] = 1.0f / lrow[i];
#pragma unroll
  for (int e4 = 0; e4 < 4; ++e4)
#pragma unroll
    for (int i = 0; i < 8; ++i) {
      const int l = l0 + wave * 16 + 8 * kh + i;
      out[((size_t)b * L_ + l) * D_ + h * E_ + e4 * 16 + lr] = o[e4][i] * lrow[i];
    }
}

// ---------------------------------------------------------------------------
extern "C" void kernel_launch(void* const* d_in, const int* in_sizes, int n_in,
                              void* d_out, int out_size, void* d_ws, size_t ws_size,
                              hipStream_t stream) {
  (void)in_sizes; (void)n_in; (void)out_size; (void)ws_size;
  const float* x = (const float*)d_in[0];       // (4, 4096, 1024) f32
  const float* w = (const float*)d_in[1];       // (1024, 3072) f32
  half_t* ws = (half_t*)d_ws;                   // 3 * 32MB f16 Q/K/V workspace
  float* out = (float*)d_out;                   // (4, 4096, 1024) f32

  dim3 g1(3 * D_ / 128, (B_ * L_) / 128);       // 24 x 128 blocks
  qkv_gemm_f16<<<g1, 256, 0, stream>>>(x, w, ws);

  dim3 g2(NW_, H_, B_);                         // 32 x 16 x 4 blocks
  local_attn_f16<<<g2, 256, 0, stream>>>(ws, out);
}
